// GroupedQueryAttention_89163521065857
// MI455X (gfx1250) — matrix operand order repaired and checked by reading, not compiled, not run on hardware
//
#include <hip/hip_runtime.h>
#include <hip/hip_bf16.h>
#include <stdint.h>

// ---------------- CDNA5 WMMA types / helpers ----------------
typedef __attribute__((ext_vector_type(16))) __bf16 v16bf;
typedef __attribute__((ext_vector_type(8)))  float  v8f;

#define WMMA_BF16(a, b, c) \
  __builtin_amdgcn_wmma_f32_16x16x32_bf16(false, (a), false, (b), (short)0, (c), false, false)

__device__ __forceinline__ unsigned short f32_to_bf16_rne(float f) {
  union { float f; unsigned int u; } cv; cv.f = f;
  unsigned int u = cv.u;
  u += 0x7FFFu + ((u >> 16) & 1u);   // round-to-nearest-even
  return (unsigned short)(u >> 16);
}

// K-permutation trick: WMMA contracts position-for-k(A) against position-for-k(B).
// Define sigma so A's 16 lane elements are CONTIGUOUS in memory (row-major, cols
// 16h..16h+15). The matching B access pattern under sigma is two contiguous
// 8-element runs per lane when B is stored N-major (transposed). Both sides use
// 128-bit loads; the contraction result is invariant to sigma.

// A fragment: 16x32 block, row-major source. Lane reads 32 contiguous bytes.
__device__ __forceinline__ v16bf load_a_sw(const unsigned short* p, int ld, int lane) {
  const int m = lane & 15, hh = lane >> 4;
  const unsigned short* r = p + (size_t)m * ld + (hh << 4);
  union { uint4 u[2]; v16bf v; } cv;
  cv.u[0] = *(const uint4*)(r);
  cv.u[1] = *(const uint4*)(r + 8);
  return cv.v;
}

// B fragment: 32x16 block whose source is stored N-major (N x K row-major).
// Lane n reads rows of src: two 16-byte runs at k-offsets 16h and 16h+8.
__device__ __forceinline__ v16bf load_b_sw_nk(const unsigned short* p, size_t ld, int lane) {
  const int n = lane & 15, hh = lane >> 4;
  const unsigned short* r = p + (size_t)n * ld + (hh << 4);
  union { uint4 u[2]; v16bf v; } cv;
  cv.u[0] = *(const uint4*)(r);        // k = 16h + e, e=0..7
  cv.u[1] = *(const uint4*)(r + 8);    // k = 16h + e, e=8..15
  return cv.v;
}

#define BB 2
#define LL 2048
#define DD 128
#define HH 16
#define GG 4

// ---------------- fp32 -> bf16 conversion (natural layout) ----------------
__global__ void cvt_f32_bf16(const float* __restrict__ in, unsigned short* __restrict__ out, int n) {
  for (int i = blockIdx.x * blockDim.x + threadIdx.x; i < n; i += gridDim.x * blockDim.x)
    out[i] = f32_to_bf16_rne(in[i]);
}

// ---------------- fp32 (K x N) -> bf16 transposed (N x K) ----------------
__global__ void cvt_f32_bf16_T(const float* __restrict__ in, unsigned short* __restrict__ out,
                               int K, int N) {
  const int total = K * N;
  for (int i = blockIdx.x * blockDim.x + threadIdx.x; i < total; i += gridDim.x * blockDim.x) {
    const int n = i / K, k = i - n * K;            // out index i = n*K + k
    out[i] = f32_to_bf16_rne(in[(size_t)k * N + n]);
  }
}

// ---------------- fused QKV GEMM + bias + RoPE ----------------
// M = B*L = 4096 rows; N tiles: 128 (q) + 32 (k) + 32 (v). One wave per 16x16 tile.
// Weights are bf16 transposed (N x 128). v is written TRANSPOSED [b][g][d][s].
__global__ void qkv_rope(const unsigned short* __restrict__ xb,
                         const unsigned short* __restrict__ WqT,
                         const unsigned short* __restrict__ WkT,
                         const unsigned short* __restrict__ WvT,
                         const float* __restrict__ bq,
                         const float* __restrict__ bk,
                         const float* __restrict__ bv,
                         unsigned short* __restrict__ qb,
                         unsigned short* __restrict__ kb,
                         unsigned short* __restrict__ vt,
                         const int* __restrict__ theta_p) {
  const int lane = threadIdx.x & 31;
  const int wid  = blockIdx.x * (blockDim.x >> 5) + (threadIdx.x >> 5);
  const int NT = 192;
  const int mt = wid / NT, nt = wid % NT;
  const int m0 = mt << 4;

  const unsigned short* WT; const float* bias;
  int col0, sect;
  if (nt < 128)      { WT = WqT; bias = bq; col0 = nt << 4;         sect = 0; }
  else if (nt < 160) { WT = WkT; bias = bk; col0 = (nt - 128) << 4; sect = 1; }
  else               { WT = WvT; bias = bv; col0 = (nt - 160) << 4; sect = 2; }

  v8f acc = {0.f, 0.f, 0.f, 0.f, 0.f, 0.f, 0.f, 0.f};
#pragma unroll
  for (int kc = 0; kc < 4; ++kc) {
    v16bf a = load_a_sw(xb + (size_t)m0 * DD + kc * 32, DD, lane);
    v16bf b = load_b_sw_nk(WT + (size_t)col0 * DD + kc * 32, DD, lane);
    acc = WMMA_BF16(a, b, acc);
  }

  const int n = lane & 15, hh = lane >> 4;
  const float bn = bias[col0 + n];
#pragma unroll
  for (int j = 0; j < 8; ++j) acc[j] += bn;

  const int col   = col0 + n;
  const int dglob = col & (DD - 1);

  if (sect != 2) {  // RoPE for q and k (interleaved even/odd pairs)
    const float theta = (float)(*theta_p);
    const int dp = dglob >> 1;
    const float invf = __powf(theta, -(float)dp / (float)(DD / 2));
#pragma unroll
    for (int j = 0; j < 8; ++j) {
      const int row = m0 + j + (hh << 3);
      const int pos = row & (LL - 1);
      const float ang = (float)pos * invf;
      const float cs = __cosf(ang), sn = __sinf(ang);
      const float self  = acc[j];
      const float other = __shfl_xor(self, 1, 32);  // partner column d^1 lives in lane^1
      acc[j] = (dglob & 1) ? (other * sn + self * cs)   // x2' = x1*sin + x2*cos
                           : (self * cs - other * sn);  // x1' = x1*cos - x2*sin
    }
  }

#pragma unroll
  for (int j = 0; j < 8; ++j) {
    const int row = m0 + j + (hh << 3);
    const int bi  = row >> 11;          // row / L
    const int l   = row & (LL - 1);
    const unsigned short v = f32_to_bf16_rne(acc[j]);
    const int hd = col >> 7, d = col & 127;
    if (sect == 0)      qb[((((size_t)bi * HH + hd) * LL + l) << 7) + d] = v;
    else if (sect == 1) kb[((((size_t)bi * GG + hd) * LL + l) << 7) + d] = v;
    else                vt[(((size_t)bi * GG + hd) * DD + d) * LL + l] = v;  // transposed
  }
}

// ---------------- flash attention: one wave per (b, h, 16-row q tile) ----------------
__global__ void __launch_bounds__(32)
attn_fa(const unsigned short* __restrict__ qb,
        const unsigned short* __restrict__ kb,
        const unsigned short* __restrict__ vt,
        unsigned short* __restrict__ ob) {
  const int lane = threadIdx.x & 31;
  const int tile = blockIdx.x;         // B*H*(L/16) = 4096
  const int lt = tile & 127;
  const int h  = (tile >> 7) & 15;
  const int bi = tile >> 11;
  const int g  = h >> 2;               // GQA: R = H/G = 4
  const int l0 = lt << 4;

  const unsigned short* qp = qb + ((((size_t)bi * HH + h) * LL + l0) << 7);
  const unsigned short* kp = kb + (((size_t)bi * GG + g) * LL << 7);
  const unsigned short* vp = vt + (((size_t)bi * GG + g) * DD) * LL;  // [d][s]

  v16bf qa[4];
#pragma unroll
  for (int kc = 0; kc < 4; ++kc) qa[kc] = load_a_sw(qp + kc * 32, DD, lane);

  const v8f zero8 = {0.f, 0.f, 0.f, 0.f, 0.f, 0.f, 0.f, 0.f};
  v8f o[8];
#pragma unroll
  for (int dc = 0; dc < 8; ++dc) o[dc] = zero8;
  float rmax[8], rsum[8];
#pragma unroll
  for (int j = 0; j < 8; ++j) { rmax[j] = -3.0e38f; rsum[j] = 0.f; }

  __shared__ __attribute__((aligned(32))) unsigned short plds[16 * 32];
  const float scale = 0.08838834764831845f;  // 1/sqrt(128)
  const int n = lane & 15, hh = lane >> 4;

  for (int s0 = 0; s0 < LL; s0 += 32) {
    if (s0 + 32 < LL) {  // prefetch next K/V blocks (global_prefetch_b8)
      __builtin_prefetch(kp + ((size_t)(s0 + 32) << 7), 0, 1);
      __builtin_prefetch(vp + (size_t)(s0 + 32), 0, 1);
    }

    // scores: two 16x16 tiles over keys [s0, s0+32). K stored [s][d] is N-major for K^T.
    v8f sc0 = zero8, sc1 = zero8;
#pragma unroll
    for (int kc = 0; kc < 4; ++kc) {
      v16bf kf = load_b_sw_nk(kp + ((size_t)s0 << 7) + kc * 32, DD, lane);
      sc0 = WMMA_BF16(qa[kc], kf, sc0);
    }
#pragma unroll
    for (int kc = 0; kc < 4; ++kc) {
      v16bf kf = load_b_sw_nk(kp + ((size_t)(s0 + 16) << 7) + kc * 32, DD, lane);
      sc1 = WMMA_BF16(qa[kc], kf, sc1);
    }

    // online softmax (row stats per half-wave: row m = j + 8*(lane>>4))
    float p0[8], p1[8], alpha[8];
#pragma unroll
    for (int j = 0; j < 8; ++j) {
      float a = sc0[j] * scale, b = sc1[j] * scale;
      float mt = fmaxf(a, b);
#pragma unroll
      for (int msk = 1; msk <= 8; msk <<= 1) mt = fmaxf(mt, __shfl_xor(mt, msk, 32));
      const float nm = fmaxf(rmax[j], mt);
      alpha[j] = __expf(rmax[j] - nm);
      a = __expf(a - nm); b = __expf(b - nm);
      float st = a + b;
#pragma unroll
      for (int msk = 1; msk <= 8; msk <<= 1) st += __shfl_xor(st, msk, 32);
      rsum[j] = rsum[j] * alpha[j] + st;
      rmax[j] = nm;
      p0[j] = a; p1[j] = b;
    }
#pragma unroll
    for (int dc = 0; dc < 8; ++dc)
#pragma unroll
      for (int j = 0; j < 8; ++j) o[dc][j] *= alpha[j];

    // P (16x32) C-layout -> LDS (row-major) -> contiguous A fragment
    __syncthreads();
#pragma unroll
    for (int j = 0; j < 8; ++j) {
      const int m = j + (hh << 3);
      plds[m * 32 + n]      = f32_to_bf16_rne(p0[j]);
      plds[m * 32 + 16 + n] = f32_to_bf16_rne(p1[j]);
    }
    __syncthreads();
    v16bf pa = load_a_sw(plds, 32, lane);

    // o += P @ V ; V^T stored [d][s] is N-major for V
#pragma unroll
    for (int dc = 0; dc < 8; ++dc) {
      v16bf vf = load_b_sw_nk(vp + (size_t)(dc * 16) * LL + s0, LL, lane);
      o[dc] = WMMA_BF16(pa, vf, o[dc]);
    }
  }

  // normalize and write [b][l][h*D] for the output GEMM
#pragma unroll
  for (int j = 0; j < 8; ++j) {
    const float inv = 1.f / rsum[j];
    const int l = l0 + j + (hh << 3);
    const size_t rowbase = ((size_t)bi * LL + l) * (HH * DD) + (size_t)h * DD;
#pragma unroll
    for (int dc = 0; dc < 8; ++dc)
      ob[rowbase + dc * 16 + n] = f32_to_bf16_rne(o[dc][j] * inv);
  }
}

// ---------------- output projection: (B*L,2048) @ (2048,128) + bo -> fp32 ----------------
__global__ void out_proj(const unsigned short* __restrict__ ob,
                         const unsigned short* __restrict__ WoT,  // 128 x 2048 (N x K)
                         const float* __restrict__ bo,
                         float* __restrict__ out) {
  const int lane = threadIdx.x & 31;
  const int wid  = blockIdx.x * (blockDim.x >> 5) + (threadIdx.x >> 5);
  const int mt = wid >> 3, ntile = wid & 7;
  const int m0 = mt << 4, n0 = ntile << 4;
  const int K = HH * DD;  // 2048

  v8f acc = {0.f, 0.f, 0.f, 0.f, 0.f, 0.f, 0.f, 0.f};
  for (int kc = 0; kc < K / 32; ++kc) {
    v16bf a = load_a_sw(ob + (size_t)m0 * K + kc * 32, K, lane);
    v16bf b = load_b_sw_nk(WoT + (size_t)n0 * K + kc * 32, K, lane);
    acc = WMMA_BF16(a, b, acc);
  }
  const int n = lane & 15, hh = lane >> 4;
  const float bn = bo[n0 + n];
#pragma unroll
  for (int j = 0; j < 8; ++j) {
    const int row = m0 + j + (hh << 3);
    out[(size_t)row * DD + n0 + n] = acc[j] + bn;
  }
}

// ---------------- host ----------------
extern "C" void kernel_launch(void* const* d_in, const int* in_sizes, int n_in,
                              void* d_out, int out_size, void* d_ws, size_t ws_size,
                              hipStream_t stream) {
  const float* x  = (const float*)d_in[0];
  const float* Wq = (const float*)d_in[1];
  const float* bq = (const float*)d_in[2];
  const float* Wk = (const float*)d_in[3];
  const float* bk = (const float*)d_in[4];
  const float* Wv = (const float*)d_in[5];
  const float* bv = (const float*)d_in[6];
  const float* Wo = (const float*)d_in[7];
  const float* bo = (const float*)d_in[8];
  const int* theta = (const int*)d_in[9];
  float* out = (float*)d_out;

  unsigned short* ws = (unsigned short*)d_ws;
  size_t off = 0;
  unsigned short* xb   = ws + off; off += (size_t)BB * LL * DD;       // x  bf16 [B*L][128]
  unsigned short* WqT  = ws + off; off += (size_t)DD * HH * DD;       // Wq^T bf16 [2048][128]
  unsigned short* WkT  = ws + off; off += (size_t)DD * GG * DD;       // [512][128]
  unsigned short* WvT  = ws + off; off += (size_t)DD * GG * DD;       // [512][128]
  unsigned short* WoT  = ws + off; off += (size_t)HH * DD * DD;       // Wo^T [128][2048]
  unsigned short* qbuf = ws + off; off += (size_t)BB * HH * LL * DD;  // q [b][h][l][d]
  unsigned short* kbuf = ws + off; off += (size_t)BB * GG * LL * DD;  // k [b][g][s][d]
  unsigned short* vbuf = ws + off; off += (size_t)BB * GG * DD * LL;  // v^T [b][g][d][s]
  unsigned short* obuf = ws + off; off += (size_t)BB * LL * HH * DD;  // attn out [b][l][h*d]
  (void)ws_size; (void)in_sizes; (void)n_in; (void)out_size;

  {
    int nEl = BB * LL * DD;
    int blocks = (nEl + 255) / 256; if (blocks > 4096) blocks = 4096;
    cvt_f32_bf16<<<blocks, 256, 0, stream>>>(x, xb, nEl);
  }
  auto cvtT = [&](const float* src, unsigned short* dst, int K, int N) {
    int nEl = K * N;
    int blocks = (nEl + 255) / 256; if (blocks > 4096) blocks = 4096;
    cvt_f32_bf16_T<<<blocks, 256, 0, stream>>>(src, dst, K, N);
  };
  cvtT(Wq, WqT, DD, HH * DD);
  cvtT(Wk, WkT, DD, GG * DD);
  cvtT(Wv, WvT, DD, GG * DD);
  cvtT(Wo, WoT, HH * DD, DD);

  // 256 M-tiles x 192 N-tiles = 49152 waves, 4 waves/block
  qkv_rope<<<12288, 128, 0, stream>>>(xb, WqT, WkT, WvT, bq, bk, bv,
                                      qbuf, kbuf, vbuf, theta);
  // one wave per (b, h, 16-row q tile)
  attn_fa<<<BB * HH * (LL / 16), 32, 0, stream>>>(qbuf, kbuf, vbuf, obuf);
  // 256 x 8 tiles = 2048 waves, 4 waves/block
  out_proj<<<512, 128, 0, stream>>>(obuf, WoT, bo, out);
}
